// KC_Avg_Embedding_76647986365123
// MI455X (gfx1250) — matrix operand order repaired and checked by reading, not compiled, not run on hardware
//
#include <hip/hip_runtime.h>

typedef __attribute__((ext_vector_type(16))) _Float16 v16h;
typedef __attribute__((ext_vector_type(8)))  _Float16 v8h;
typedef __attribute__((ext_vector_type(4)))  _Float16 v4h;
typedef __attribute__((ext_vector_type(8)))  float    v8f;
typedef __attribute__((ext_vector_type(4)))  float    f32x4;
typedef __attribute__((ext_vector_type(4)))  unsigned int u32x4;

union Frag16 {
  v16h  h;
  u32x4 u[2];
};

#define KDIM     1024   // V
#define DDIM     128    // D
#define KCHUNK   32     // K per WMMA step
#define MROWS    64     // rows per workgroup
#define NTILES   4      // 16-row M tiles per workgroup
#define NTHREADS 288    // 8 compute waves + 1 rowsum wave

// ---- one-time E transpose + f32->f16 downconvert: Et[n][k] = (f16)E[k][n] ----
// 0.5 MB read / 0.25 MB written once; leaves Et L2-resident for the main kernel.
__global__ __launch_bounds__(256)
void transpose_E_f16(const float* __restrict__ E, _Float16* __restrict__ Et)
{
  const int n = blockIdx.x;    // 0..DDIM-1
  const int t = threadIdx.x;   // handles k = 4t .. 4t+3
  v4h pk;
  #pragma unroll
  for (int i = 0; i < 4; ++i)
    pk[i] = (_Float16)E[(size_t)(4 * t + i) * DDIM + n];
  *(v4h*)&Et[(size_t)n * KDIM + 4 * t] = pk;   // coalesced b64 stores
}

template <bool USE_ET>
__global__ __launch_bounds__(NTHREADS)
void kc_avg_embedding_wmma(const float* __restrict__ A,     // [M, K] multi-hot 0/1
                           const float* __restrict__ E,     // [K, D] f32
                           const _Float16* __restrict__ Et, // [D, K] f16 (if USE_ET)
                           float* __restrict__ O,           // [M, D]
                           int M)
{
  // double-buffered f16 staging of the 64x32 A chunk (2 x 4 KB) + rowsum broadcast
  __shared__ _Float16 abuf[2][MROWS * KCHUNK];
  __shared__ float    sums[MROWS];

  const int t    = threadIdx.x;
  const int lane = t & 31;
  const int wave = t >> 5;        // 0..7 compute (one 16-col N tile each), 8 = rowsum
  const int half = lane >> 4;     // 0 = lanes 0-15, 1 = lanes 16-31
  const int l15  = lane & 15;

  const int row0 = blockIdx.x * MROWS;

  // all-ones B fragment for the rowsum wave
  v16h b_ones;
  #pragma unroll
  for (int i = 0; i < 16; ++i) b_ones[i] = (_Float16)1.0f;

  v8f c[NTILES];
  #pragma unroll
  for (int mt = 0; mt < NTILES; ++mt)
    c[mt] = (v8f){0.f,0.f,0.f,0.f,0.f,0.f,0.f,0.f};

  // staging (threads 0..255): thread t converts 8 contiguous f32 of the 64x32 chunk
  const int srow = t >> 2;          // row within WG tile (0..63)
  const int skk  = (t & 3) * 8;     // k offset within chunk (0,8,16,24)
  const int ncol = wave * 16 + l15; // output column for compute waves

  const float* pa = &A[(size_t)(row0 + srow) * KDIM + skk];  // used only when t < 256

  for (int kc = 0; kc < KDIM / KCHUNK; ++kc) {
    const int kbase = kc * KCHUNK;
    _Float16* buf = abuf[kc & 1];

    // ---- stage A chunk (fp32 -> fp16) into LDS: 2x b128 load, 1x ds_store_b128 ----
    if (t < 256) {
      f32x4 x0 = *(const f32x4*)(pa + kbase);
      f32x4 x1 = *(const f32x4*)(pa + kbase + 4);
      v8h pk;
      #pragma unroll
      for (int i = 0; i < 4; ++i) {
        pk[i]     = (_Float16)x0[i];
        pk[4 + i] = (_Float16)x1[i];
      }
      *(v8h*)&buf[srow * KCHUNK + skk] = pk;

      // keep the 105 MB A stream flowing: prefetch two chunks ahead
      if (kc + 2 < KDIM / KCHUNK)
        __builtin_prefetch((const void*)(pa + kbase + 2 * KCHUNK), 0, 1);
    }
    __syncthreads();

    if (wave < 8) {
      // B fragment once per chunk, reused across all 4 M tiles.
      // v16h elements 0..15 = K pairs (2r,2r+1); lanes 0-15 cover K 0..15,
      // lanes 16-31 cover K 16..31 of this chunk.
      Frag16 fb;
      if (USE_ET) {
        // contiguous 32 B per lane from the pre-transposed f16 copy
        const _Float16* pb = &Et[(size_t)ncol * KDIM + kbase + half * 16];
        fb.u[0] = *(const u32x4*)(pb);       // K half*16 .. +7
        fb.u[1] = *(const u32x4*)(pb + 8);   // K half*16+8 .. +15
      } else {
        const float* pb = &E[(size_t)(kbase + half * 16) * DDIM + ncol];
        #pragma unroll
        for (int r = 0; r < 8; ++r) {
          fb.h[2 * r]     = (_Float16)pb[(size_t)(2 * r)     * DDIM];
          fb.h[2 * r + 1] = (_Float16)pb[(size_t)(2 * r + 1) * DDIM];
        }
      }
      #pragma unroll
      for (int mt = 0; mt < NTILES; ++mt) {
        Frag16 fa;   // A fragment, K half*16..+15 contiguous (matches B): two ds_load_b128
        const _Float16* p = &buf[(mt * 16 + l15) * KCHUNK + half * 16];
        fa.u[0] = *(const u32x4*)(p);
        fa.u[1] = *(const u32x4*)(p + 8);
        c[mt] = __builtin_amdgcn_wmma_f32_16x16x32_f16(false, fa.h, false, fb.h,
                                                       (short)0, c[mt], false, false);
      }
    } else {
      // dedicated rowsum wave: same A rows, k half*8..+7 and 16+half*8..+7, x all-ones B
      #pragma unroll
      for (int mt = 0; mt < NTILES; ++mt) {
        Frag16 fa;
        const _Float16* p = &buf[(mt * 16 + l15) * KCHUNK + half * 8];
        fa.u[0] = *(const u32x4*)(p);
        fa.u[1] = *(const u32x4*)(p + 16);
        c[mt] = __builtin_amdgcn_wmma_f32_16x16x32_f16(false, fa.h, false, b_ones,
                                                       (short)0, c[mt], false, false);
      }
    }
  }

  // ---- rowsum wave publishes per-row counts (accumulator VGPR g -> row g+8*half) ----
  if (wave == 8 && l15 == 0) {
    #pragma unroll
    for (int mt = 0; mt < NTILES; ++mt)
      #pragma unroll
      for (int g = 0; g < 8; ++g)
        sums[mt * 16 + g + 8 * half] = c[mt][g];
  }
  __syncthreads();

  // ---- epilogue: divide by clipped count, store 4 x 16x16 f32 tiles ----
  if (wave < 8) {
    #pragma unroll
    for (int mt = 0; mt < NTILES; ++mt) {
      const size_t obase = (size_t)(row0 + mt * 16 + half * 8) * DDIM + ncol;
      #pragma unroll
      for (int g = 0; g < 8; ++g) {
        float denom = fmaxf(sums[mt * 16 + g + 8 * half], 1.0f);
        O[obase + (size_t)g * DDIM] = c[mt][g] / denom;
      }
    }
  }
}

extern "C" void kernel_launch(void* const* d_in, const int* in_sizes, int n_in,
                              void* d_out, int out_size, void* d_ws, size_t ws_size,
                              hipStream_t stream) {
  const float* A = (const float*)d_in[0];   // batch_vectors [B,S,V] fp32
  const float* E = (const float*)d_in[1];   // embedding_matrix [V,D] fp32
  float* O = (float*)d_out;                 // [B,S,D] fp32

  const int M = in_sizes[0] / KDIM;         // B*S = 25600 (multiple of MROWS)
  dim3 grid(M / MROWS), block(NTHREADS);

  const size_t etBytes = (size_t)KDIM * DDIM * sizeof(_Float16);  // 256 KB
  if (ws_size >= etBytes) {
    _Float16* Et = (_Float16*)d_ws;
    hipLaunchKernelGGL(transpose_E_f16, dim3(DDIM), dim3(256), 0, stream, E, Et);
    hipLaunchKernelGGL((kc_avg_embedding_wmma<true>), grid, block, 0, stream,
                       A, E, (const _Float16*)Et, O, M);
  } else {
    hipLaunchKernelGGL((kc_avg_embedding_wmma<false>), grid, block, 0, stream,
                       A, E, (const _Float16*)nullptr, O, M);
  }
}
